// TriangleMultiplication_52948356825355
// MI455X (gfx1250) — matrix operand order repaired and checked by reading, not compiled, not run on hardware
//
#include <hip/hip_runtime.h>
#include <hip/hip_bf16.h>
#include <math.h>

#define NT   512
#define CC   128
#define EPSV 1e-5f

typedef __attribute__((ext_vector_type(16))) __bf16 bf16x16;
typedef __attribute__((ext_vector_type(8)))  __bf16 bf16x8;
typedef __attribute__((ext_vector_type(8)))  float  f32x8;
typedef __attribute__((ext_vector_type(4)))  int    v4i;

__device__ __forceinline__ bf16x16 bcat(bf16x8 a, bf16x8 b){
  bf16x16 r;
#pragma unroll
  for (int i=0;i<8;++i){ r[i]=a[i]; r[i+8]=b[i]; }
  return r;
}

__device__ __forceinline__ f32x8 wmma_bf16(bf16x16 a, bf16x16 b, f32x8 c){
  return __builtin_amdgcn_wmma_f32_16x16x32_bf16(false, a, false, b, (short)0, c, false, false);
}

__device__ __forceinline__ float sigmoidf(float x){ return 1.0f/(1.0f + __expf(-x)); }

// ---- CDNA5 async global->LDS (ASYNCcnt) with safe fallback -------------------
#if defined(__gfx1250__) && __has_builtin(__builtin_amdgcn_global_load_async_to_lds_b128)
#define HAVE_ASYNC_LDS 1
typedef __attribute__((address_space(1))) v4i* g_v4i_ptr;
typedef __attribute__((address_space(3))) v4i* l_v4i_ptr;
__device__ __forceinline__ void async_b128(void* lds, const void* g){
  __builtin_amdgcn_global_load_async_to_lds_b128((g_v4i_ptr)g, (l_v4i_ptr)lds, 0, 0);
}
template <int N>
__device__ __forceinline__ void wait_async(){
#if __has_builtin(__builtin_amdgcn_s_wait_asynccnt)
  __builtin_amdgcn_s_wait_asynccnt(N);
#else
  asm volatile("s_wait_asynccnt %0" :: "i"(N) : "memory");
#endif
}
#else
#define HAVE_ASYNC_LDS 0
__device__ __forceinline__ void async_b128(void* lds, const void* g){
  *(bf16x8*)lds = *(const bf16x8*)g;   // synchronous fallback
}
template <int N>
__device__ __forceinline__ void wait_async(){}
#endif

// ---------------- Phase 0: weight conversion to bf16 ----------------
__global__ void k_cvt_bf16(const float* __restrict__ src, __bf16* __restrict__ dst, int n){
  int i = blockIdx.x*blockDim.x + threadIdx.x;
  if (i < n) dst[i] = (__bf16)src[i];
}

// ---------------- Phase 1: LN1 + proj/gate/glin GEMMs (fused) ----------------
// grid (512, 32): x = r1 (=k of einsum), y = tile of 16 r2 tokens. block = 256 (8 waves).
__global__ __launch_bounds__(256) void k_ln_proj(
    const float* __restrict__ pair, const float* __restrict__ mask,
    const float* __restrict__ ln1w, const float* __restrict__ ln1b,
    const __bf16* __restrict__ wp, const __bf16* __restrict__ wg,
    const __bf16* __restrict__ wl,
    __bf16* __restrict__ a_st,   // [c][j][k]
    __bf16* __restrict__ b_st,   // [c][i][k]
    __bf16* __restrict__ gsig)   // [tok][c]
{
  __shared__ alignas(16) __bf16 xa[16][136];
  __shared__ float redS[16][16];
  __shared__ float redQ[16][16];
  __shared__ float mu_s[16], rs_s[16], mk_s[16];

  const int r1   = blockIdx.x;
  const int r2_0 = blockIdx.y * 16;
  const int tid  = threadIdx.x;
  const int m    = tid >> 4;
  const int g    = tid & 15;

  const float* prow = pair + ((size_t)r1*NT + r2_0 + m) * CC + g*8;
  float v[8]; float s=0.f, q=0.f;
#pragma unroll
  for (int i=0;i<8;++i){ v[i]=prow[i]; s+=v[i]; q+=v[i]*v[i]; }
  redS[m][g]=s; redQ[m][g]=q;
  __syncthreads();
  if (g==0){
    float ss=0.f, qq=0.f;
#pragma unroll
    for (int i=0;i<16;++i){ ss+=redS[m][i]; qq+=redQ[m][i]; }
    float mu  = ss * (1.0f/CC);
    float var = qq * (1.0f/CC) - mu*mu;
    mu_s[m]=mu; rs_s[m]=rsqrtf(var + EPSV);
    mk_s[m]=mask[(size_t)r1*NT + r2_0 + m];
  }
  __syncthreads();
  {
    float mu = mu_s[m], rs = rs_s[m];
#pragma unroll
    for (int i=0;i<8;++i){
      int c = g*8+i;
      xa[m][c] = (__bf16)((v[i]-mu)*rs*ln1w[c] + ln1b[c]);
    }
  }
  __syncthreads();

  const int wid  = tid >> 5;
  const int lane = tid & 31;
  const int row  = lane & 15;
  const int ka   = (lane < 16) ? 0 : 16;
  const int kb   = (lane < 16) ? 0 : 16;

  bf16x16 afr[4];
#pragma unroll
  for (int ks=0; ks<4; ++ks)
    afr[ks] = bcat(*(const bf16x8*)&xa[row][ks*32 + ka],
                   *(const bf16x8*)&xa[row][ks*32 + ka + 8]);

#pragma unroll
  for (int uu=0; uu<2; ++uu){
    const int o0 = (wid + uu*8) * 16;
    f32x8 accP = {}; f32x8 accG = {};
#pragma unroll
    for (int ks=0; ks<4; ++ks){
      const __bf16* bp = wp + (size_t)(o0 + row)*CC + ks*32 + kb;
      const __bf16* bg = wg + (size_t)(o0 + row)*CC + ks*32 + kb;
      bf16x16 B1 = bcat(*(const bf16x8*)bp, *(const bf16x8*)(bp+8));
      bf16x16 B2 = bcat(*(const bf16x8*)bg, *(const bf16x8*)(bg+8));
      accP = wmma_bf16(afr[ks], B1, accP);
      accG = wmma_bf16(afr[ks], B2, accG);
    }
    const int o  = o0 + row;
    const int mb = (lane<16)?0:8;
    __bf16* dst  = (o & 1) ? b_st : a_st;
    const size_t cbase = (size_t)(o>>1)*((size_t)NT*NT) + (size_t)r1;
#pragma unroll
    for (int r=0;r<8;++r){
      int mm = mb + r;
      float val = accP[r] * mk_s[mm] * sigmoidf(accG[r]);
      dst[cbase + (size_t)(r2_0+mm)*NT] = (__bf16)val;
    }
  }
  {
    const int o0 = wid*16;
    f32x8 accL = {};
#pragma unroll
    for (int ks=0; ks<4; ++ks){
      const __bf16* bl = wl + (size_t)(o0+row)*CC + ks*32 + kb;
      bf16x16 B = bcat(*(const bf16x8*)bl, *(const bf16x8*)(bl+8));
      accL = wmma_bf16(afr[ks], B, accL);
    }
    const int o  = o0 + row;
    const int mb = (lane<16)?0:8;
#pragma unroll
    for (int r=0;r<8;++r){
      size_t tok = (size_t)r1*NT + r2_0 + mb + r;
      gsig[tok*CC + o] = (__bf16)sigmoidf(accL[r]);
    }
  }
}

// ---------------- Phase 2: triangle einsum = per-channel 512x512x512 GEMM ----------------
// out[i,j] = sum_k b[i][k] * a[j][k].  grid (8,2,128): i-tile(64), j-tile(256), c.
// Double-buffered LDS fed by async global->LDS b128; 8 waves in 2x4, 32x64 per wave.
__global__ __launch_bounds__(256) void k_tri(
    const __bf16* __restrict__ a_st, const __bf16* __restrict__ b_st,
    float* __restrict__ otri)        // [c][i][j]
{
  __shared__ alignas(16) __bf16 Asl[2][64][40];
  __shared__ alignas(16) __bf16 Bsl[2][256][40];
  const int i0 = blockIdx.x * 64;
  const int j0 = blockIdx.y * 256;
  const int c  = blockIdx.z;
  const __bf16* Ac = b_st + (size_t)c*((size_t)NT*NT);  // rows i, cols k
  const __bf16* Bc = a_st + (size_t)c*((size_t)NT*NT);  // rows j, cols k
  const int tid = threadIdx.x;
  const int wid = tid >> 5, lane = tid & 31;
  const int wm = wid >> 2, wn = wid & 3;    // 2 x 4 wave grid
  const int row = lane & 15;
  const int ka  = (lane<16)?0:16;
  const int kb  = (lane<16)?0:16;

  // slab-load assignments (16B chunks, coalesced)
  const int ra = tid >> 2, ca_ = (tid & 3)*8;   // A: 64 rows x 32 halfs, 1 chunk/thread

  f32x8 acc[2][4] = {};

  const int NSTEP = NT/32;
  int buf = 0;

  // issue stage ks into buffer b
  auto issue = [&](int ks, int b){
    const int k0 = ks*32;
    async_b128(&Asl[b][ra][ca_], Ac + (size_t)(i0+ra)*NT + k0 + ca_);
#pragma unroll
    for (int t=0;t<4;++t){                      // B: 256 rows, 4 chunks/thread
      int cid = tid + t*256;
      int r = cid >> 2, ch = (cid & 3)*8;
      async_b128(&Bsl[b][r][ch], Bc + (size_t)(j0+r)*NT + k0 + ch);
    }
  };

  issue(0, 0);
  for (int s=0; s<NSTEP; ++s){
    if (s+1 < NSTEP) issue(s+1, buf^1);
    // 5 async b128 issued per thread per stage; leave next stage in flight
    if (s+1 < NSTEP) wait_async<5>(); else wait_async<0>();
    __syncthreads();                           // stage s resident for all waves

    bf16x16 afr[2], bfr[4];
#pragma unroll
    for (int mt=0; mt<2; ++mt)
      afr[mt] = bcat(*(const bf16x8*)&Asl[buf][wm*32+mt*16+row][ka],
                     *(const bf16x8*)&Asl[buf][wm*32+mt*16+row][ka+8]);
#pragma unroll
    for (int nt=0; nt<4; ++nt)
      bfr[nt] = bcat(*(const bf16x8*)&Bsl[buf][wn*64+nt*16+row][kb],
                     *(const bf16x8*)&Bsl[buf][wn*64+nt*16+row][kb+8]);
#pragma unroll
    for (int mt=0; mt<2; ++mt)
#pragma unroll
      for (int nt=0; nt<4; ++nt)
        acc[mt][nt] = wmma_bf16(afr[mt], bfr[nt], acc[mt][nt]);

    buf ^= 1;
    if (s+1 < NSTEP) __syncthreads();          // all waves done with old buffer
  }

  float* oc = otri + (size_t)c*((size_t)NT*NT);
  const int mbo = (lane<16)?0:8;
#pragma unroll
  for (int mt=0; mt<2; ++mt)
#pragma unroll
    for (int nt=0; nt<4; ++nt){
      int jcol = j0 + wn*64 + nt*16 + row;
      int irow = i0 + wm*32 + mt*16 + mbo;
#pragma unroll
      for (int r=0;r<8;++r)
        oc[(size_t)(irow+r)*NT + jcol] = acc[mt][nt][r];
    }
}

// ---------------- Phase 3: LN2 + W_out GEMM + gating ----------------
// grid (512, 8): i, j-tile of 64. block = 256.
__global__ __launch_bounds__(256) void k_final(
    const float* __restrict__ otri,
    const float* __restrict__ ln2w, const float* __restrict__ ln2b,
    const __bf16* __restrict__ wo, const __bf16* __restrict__ gsig,
    float* __restrict__ out)
{
  __shared__ float  yl[64][129];
  __shared__ alignas(16) __bf16 xb[64][136];
  __shared__ float  ps[64][4][2];
  __shared__ float  mu_s[64], rs_s[64];

  const int i   = blockIdx.x;
  const int j0  = blockIdx.y * 64;
  const int tid = threadIdx.x;

#pragma unroll 1
  for (int it=0; it<32; ++it){
    int c = it*4 + (tid >> 6);
    int j = tid & 63;
    yl[j][c] = otri[(size_t)c*((size_t)NT*NT) + (size_t)i*NT + j0 + j];
  }
  __syncthreads();
  {
    int j = tid >> 2, part = tid & 3;
    float s=0.f, q=0.f;
#pragma unroll
    for (int cc=0; cc<32; ++cc){
      float v = yl[j][part*32+cc];
      s += v; q += v*v;
    }
    ps[j][part][0]=s; ps[j][part][1]=q;
  }
  __syncthreads();
  if (tid < 64){
    int j = tid;
    float s = ps[j][0][0]+ps[j][1][0]+ps[j][2][0]+ps[j][3][0];
    float q = ps[j][0][1]+ps[j][1][1]+ps[j][2][1]+ps[j][3][1];
    float mu  = s*(1.0f/CC);
    float var = q*(1.0f/CC) - mu*mu;
    mu_s[j]=mu; rs_s[j]=rsqrtf(var+EPSV);
  }
  __syncthreads();
  {
    int j = tid >> 2, c0 = (tid & 3)*32;
    float mu = mu_s[j], rs = rs_s[j];
#pragma unroll
    for (int cc=0; cc<32; ++cc){
      int c = c0+cc;
      xb[j][c] = (__bf16)((yl[j][c]-mu)*rs*ln2w[c] + ln2b[c]);
    }
  }
  __syncthreads();

  const int wid  = tid >> 5, lane = tid & 31;
  const int row  = lane & 15;
  const int ka   = (lane<16)?0:16;
  const int kb   = (lane<16)?0:16;
  const int o    = wid*16 + row;
  const int mbo  = (lane<16)?0:8;

#pragma unroll 1
  for (int mt=0; mt<4; ++mt){
    f32x8 acc = {};
#pragma unroll
    for (int ks=0; ks<4; ++ks){
      bf16x16 A = bcat(*(const bf16x8*)&xb[mt*16+row][ks*32+ka],
                       *(const bf16x8*)&xb[mt*16+row][ks*32+ka+8]);
      const __bf16* wr = wo + (size_t)o*CC + ks*32 + kb;
      bf16x16 B = bcat(*(const bf16x8*)wr, *(const bf16x8*)(wr+8));
      acc = wmma_bf16(A, B, acc);
    }
#pragma unroll
    for (int r=0;r<8;++r){
      size_t tok = (size_t)i*NT + (j0 + mt*16 + mbo + r);
      float gv = (float)gsig[tok*CC + o];
      out[tok*CC + o] = acc[r] * gv;
    }
  }
}

// ---------------- launcher ----------------
extern "C" void kernel_launch(void* const* d_in, const int* in_sizes, int n_in,
                              void* d_out, int out_size, void* d_ws, size_t ws_size,
                              hipStream_t stream) {
  const float* pair  = (const float*)d_in[0];
  const float* mask  = (const float*)d_in[1];
  const float* ln1w  = (const float*)d_in[2];
  const float* ln1b  = (const float*)d_in[3];
  const float* ln2w  = (const float*)d_in[4];
  const float* ln2b  = (const float*)d_in[5];
  const float* Wproj = (const float*)d_in[6];
  const float* Wgate = (const float*)d_in[7];
  const float* Wout  = (const float*)d_in[8];
  const float* Wglin = (const float*)d_in[9];
  float* out = (float*)d_out;

  char* ws = (char*)d_ws;
  const size_t SZ_CH = (size_t)CC * NT * NT;
  __bf16* wp_bf = (__bf16*)(ws + 0);
  __bf16* wg_bf = (__bf16*)(ws + 65536);
  __bf16* wl_bf = (__bf16*)(ws + 131072);
  __bf16* wo_bf = (__bf16*)(ws + 163840);
  __bf16* a_st  = (__bf16*)(ws + 196608);
  __bf16* b_st  = (__bf16*)(ws + 196608 + 2ull*SZ_CH);
  __bf16* gsig  = (__bf16*)(ws + 196608 + 4ull*SZ_CH);
  float*  otri  = (float*) (ws + 196608 + 6ull*SZ_CH);

  k_cvt_bf16<<<128, 256, 0, stream>>>(Wproj, wp_bf, 32768);
  k_cvt_bf16<<<128, 256, 0, stream>>>(Wgate, wg_bf, 32768);
  k_cvt_bf16<<<64,  256, 0, stream>>>(Wglin, wl_bf, 16384);
  k_cvt_bf16<<<64,  256, 0, stream>>>(Wout,  wo_bf, 16384);

  k_ln_proj<<<dim3(512,32), 256, 0, stream>>>(pair, mask, ln1w, ln1b,
                                              wp_bf, wg_bf, wl_bf,
                                              a_st, b_st, gsig);
  k_tri<<<dim3(8,2,128), 256, 0, stream>>>(a_st, b_st, otri);
  k_final<<<dim3(512,8), 256, 0, stream>>>(otri, ln2w, ln2b, wo_bf, gsig, out);
}
